// Net_51196010168661
// MI455X (gfx1250) — matrix operand order repaired and checked by reading, not compiled, not run on hardware
//
#include <hip/hip_runtime.h>
#include <hip/hip_bf16.h>
#include <math.h>

typedef __attribute__((ext_vector_type(16))) _Float16 v16h;
typedef __attribute__((ext_vector_type(8)))  _Float16 v8h;
typedef __attribute__((ext_vector_type(8)))  float    v8f;

#define NPTS 8192
#define RAD2 0.0625f
#define NCENT 32
#define KNBR 32
#define BATCH 128
#define BNEPS 1e-5f

// ---------------- workspace layout (in float units; f16 buffers use halves) ---
constexpr size_t OFF_CENT = 0;                        // 128*2*32 f32
constexpr size_t OFF_G    = OFF_CENT + 8192;          // 128*2*32*32 f32
constexpr size_t OFF_H1T  = OFF_G    + 262144;        // 128*1024*64 f16  (4194304 f)
constexpr size_t OFF_H2T  = OFF_H1T  + 4194304;       // 128*1024*128 f16 (8388608 f)
constexpr size_t OFF_POOL = OFF_H2T  + 8388608;       // 128*256*32 f32
constexpr size_t OFF_FCIN = OFF_POOL + 1048576;       // 128*8192 f16 (524288 f)
constexpr size_t OFF_FC1  = OFF_FCIN + 524288;        // 128*1024 f32
constexpr size_t OFF_FC1P = OFF_FC1  + 131072;        // 128*1024 f16 (65536 f)
constexpr size_t OFF_FC2  = OFF_FC1P + 65536;         // 128*256 f32
constexpr size_t OFF_W2   = OFF_FC2  + 32768;         // 128*64 f16   (4096 f)
constexpr size_t OFF_W3   = OFF_W2   + 4096;          // 256*128 f16  (16384 f)
constexpr size_t OFF_WF2  = OFF_W3   + 16384;         // 256*1024 f16 (131072 f)
constexpr size_t OFF_WF1  = OFF_WF2  + 131072;        // 1024*8192 f16 (4194304 f)
constexpr size_t OFF_BN1M = OFF_WF1  + 4194304;       // 64
constexpr size_t OFF_BN1I = OFF_BN1M + 64;
constexpr size_t OFF_BN2M = OFF_BN1I + 64;            // 128
constexpr size_t OFF_BN2I = OFF_BN2M + 128;
constexpr size_t OFF_BN3S = OFF_BN2I + 128;           // 512 (sum|sumsq)
constexpr size_t OFF_BN3M = OFF_BN3S + 512;           // 256
constexpr size_t OFF_BN3I = OFF_BN3M + 256;
constexpr size_t OFF_BN4M = OFF_BN3I + 256;           // 1024
constexpr size_t OFF_BN4I = OFF_BN4M + 1024;
constexpr size_t OFF_BN5M = OFF_BN4I + 1024;          // 256
constexpr size_t OFF_BN5I = OFF_BN5M + 256;

// ---------------- init ------------------------------------------------------
__global__ void init_kernel(float* pooled, float* bn3sums) {
    int i = blockIdx.x * 256 + threadIdx.x;
    if (i < BATCH * 256 * NCENT) pooled[i] = -3.402823466e38f;
    if (i < 512) bn3sums[i] = 0.f;
}

// ---------------- FPS: one block per batch, dist resident in LDS ------------
__global__ void fps_kernel(const float* __restrict__ x, float* __restrict__ cent) {
    __shared__ float dist[NPTS];
    __shared__ float red[256];
    __shared__ int   redi[256];
    __shared__ float curc[2];
    int b = blockIdx.x, t = threadIdx.x;
    const float* x0 = x + (long)b * 2 * NPTS;
    const float* x1 = x0 + NPTS;
    float c0x = x0[0], c0y = x1[0];
    for (int i = t; i < NPTS; i += 256) {
        float dx = x0[i] - c0x, dy = x1[i] - c0y;
        dist[i] = dx * dx + dy * dy;
    }
    if (t == 0) { cent[b * 64 + 0] = c0x; cent[b * 64 + 32] = c0y; }
    __syncthreads();
    for (int it = 1; it < NCENT; ++it) {
        float best = -1.f; int bi = 0x7fffffff;
        for (int i = t; i < NPTS; i += 256) {
            float d = dist[i];
            if (d > best) { best = d; bi = i; }
        }
        red[t] = best; redi[t] = bi;
        __syncthreads();
        for (int s = 128; s > 0; s >>= 1) {
            if (t < s) {
                if (red[t + s] > red[t] ||
                    (red[t + s] == red[t] && redi[t + s] < redi[t])) {
                    red[t] = red[t + s]; redi[t] = redi[t + s];
                }
            }
            __syncthreads();
        }
        if (t == 0) {
            int j = redi[0];
            curc[0] = x0[j]; curc[1] = x1[j];
            cent[b * 64 + it]      = x0[j];
            cent[b * 64 + 32 + it] = x1[j];
        }
        __syncthreads();
        float cx = curc[0], cy = curc[1];
        for (int i = t; i < NPTS; i += 256) {
            float dx = x0[i] - cx, dy = x1[i] - cy;
            float d = dx * dx + dy * dy;
            if (d < dist[i]) dist[i] = d;
        }
        __syncthreads();
    }
}

// ---------------- grouping: one wave per (b,c); ballot-ordered collect ------
__global__ void group_kernel(const float* __restrict__ x,
                             const float* __restrict__ cent,
                             float* __restrict__ g) {
    int c = blockIdx.x, b = blockIdx.y, lane = threadIdx.x;
    const float* x0 = x + (long)b * 2 * NPTS;
    const float* x1 = x0 + NPTS;
    float cx = cent[b * 64 + c], cy = cent[b * 64 + 32 + c];
    float* g0 = g + ((long)b * 2 + 0) * (NCENT * KNBR) + c * KNBR;
    float* g1 = g + ((long)b * 2 + 1) * (NCENT * KNBR) + c * KNBR;
    __shared__ float firstp[2];
    int cnt = 0;
    for (int base = 0; base < NPTS && cnt < KNBR; base += 32) {
        int i = base + lane;
        float px = x0[i], py = x1[i];
        float dx = px - cx, dy = py - cy;
        bool in = (dx * dx + dy * dy) <= RAD2;
        unsigned mask = (unsigned)__ballot(in);
        int pre = __popc(mask & ((1u << lane) - 1u));
        int slot = cnt + pre;
        if (in && slot < KNBR) {
            g0[slot] = px; g1[slot] = py;
            if (slot == 0) { firstp[0] = px; firstp[1] = py; }
        }
        cnt += __popc(mask);
    }
    __syncthreads();
    float fx, fy;
    if (cnt > 0) { fx = firstp[0]; fy = firstp[1]; }
    else         { fx = x0[NPTS - 1]; fy = x1[NPTS - 1]; }   // JAX OOB clamp
    if (cnt < KNBR)
        for (int s = cnt + lane; s < KNBR; s += 32) { g0[s] = fx; g1[s] = fy; }
}

// ------- conv1 (K=2): write pre-BN f16 TRANSPOSED [b][ck][64] ---------------
__global__ void conv1t_kernel(const float* __restrict__ g,
                              const float* __restrict__ w,
                              const float* __restrict__ bias,
                              _Float16* __restrict__ h1t) {
    int ck = blockIdx.x * 256 + threadIdx.x;
    int b = blockIdx.y;
    float g0 = g[((long)b * 2 + 0) * 1024 + ck];
    float g1 = g[((long)b * 2 + 1) * 1024 + ck];
    _Float16* dst = h1t + ((long)b * 1024 + ck) * 64;
#pragma unroll
    for (int o = 0; o < 64; ++o)
        dst[o] = (_Float16)(w[2 * o] * g0 + w[2 * o + 1] * g1 + bias[o]);
}

// ------- per-channel mean/invstd over f16 buffer [i][ch], ch stride = nch ---
__global__ void stats16_kernel(const _Float16* __restrict__ h,
                               float* __restrict__ mean, float* __restrict__ inv,
                               int count, int stride) {
    int ch = blockIdx.x, t = threadIdx.x;
    float s = 0.f, ss = 0.f;
    for (int i = t; i < count; i += 256) {
        float v = (float)h[(long)i * stride + ch];
        s += v; ss += v * v;
    }
    __shared__ float rs[256], rss[256];
    rs[t] = s; rss[t] = ss;
    __syncthreads();
    for (int k = 128; k > 0; k >>= 1) {
        if (t < k) { rs[t] += rs[t + k]; rss[t] += rss[t + k]; }
        __syncthreads();
    }
    if (t == 0) {
        float m = rs[0] / (float)count;
        float v = rss[0] / (float)count - m * m;
        mean[ch] = m;
        inv[ch] = rsqrtf(fmaxf(v, 0.f) + BNEPS);
    }
}

// ------- per-channel mean/invstd over f32 buffer (fc layers) ----------------
__global__ void stats32_kernel(const float* __restrict__ h,
                               float* __restrict__ mean, float* __restrict__ inv,
                               int count, int stride) {
    int ch = blockIdx.x, t = threadIdx.x;
    float s = 0.f, ss = 0.f;
    for (int i = t; i < count; i += 256) {
        float v = h[(long)i * stride + ch];
        s += v; ss += v * v;
    }
    __shared__ float rs[256], rss[256];
    rs[t] = s; rss[t] = ss;
    __syncthreads();
    for (int k = 128; k > 0; k >>= 1) {
        if (t < k) { rs[t] += rs[t + k]; rss[t] += rss[t + k]; }
        __syncthreads();
    }
    if (t == 0) {
        float m = rs[0] / (float)count;
        float v = rss[0] / (float)count - m * m;
        mean[ch] = m;
        inv[ch] = rsqrtf(fmaxf(v, 0.f) + BNEPS);
    }
}

// ------- in-place BN+ReLU on f16 buffer, channel = i & mask -----------------
__global__ void bnrelu16_kernel(_Float16* __restrict__ h,
                                const float* __restrict__ m, const float* __restrict__ inv,
                                const float* __restrict__ g, const float* __restrict__ b_,
                                int mask, long total) {
    long i = (long)blockIdx.x * 256 + threadIdx.x;
    if (i < total) {
        int ch = (int)i & mask;
        float v = (float)h[i];
        h[i] = (_Float16)fmaxf(0.f, (v - m[ch]) * inv[ch] * g[ch] + b_[ch]);
    }
}

// ------- plain f32 -> f16 conversion (weights) ------------------------------
__global__ void cvt_kernel(const float* __restrict__ src, _Float16* __restrict__ dst, long n) {
    long i = (long)blockIdx.x * 256 + threadIdx.x;
    if (i < n) dst[i] = (_Float16)src[i];
}

// ------- f32 -> f16 with BN+ReLU (fc1 activations) --------------------------
__global__ void cvtbn_kernel(const float* __restrict__ src, _Float16* __restrict__ dst,
                             const float* __restrict__ m, const float* __restrict__ inv,
                             const float* __restrict__ g, const float* __restrict__ b_,
                             int mask, long total) {
    long i = (long)blockIdx.x * 256 + threadIdx.x;
    if (i < total) {
        int ch = (int)i & mask;
        float v = src[i];
        dst[i] = (_Float16)fmaxf(0.f, (v - m[ch]) * inv[ch] * g[ch] + b_[ch]);
    }
}

__device__ inline void atomicMaxF(float* addr, float val) {
    if (val >= 0.f) atomicMax((int*)addr, __float_as_int(val));
    else            atomicMin((unsigned int*)addr, __float_as_uint(val));
}

// ---------------- pure-f16 WMMA GEMM ----------------------------------------
// C[M,N] = A16[M,K] x B16[N,K]^T   (both K-contiguous; 2x b128 per fragment)
// Wave tile = 16 x 64 (4 accumulators share one A fragment).
// EPI 0: f32 store C[M,N] + bias[n]            (fc1, fc2)
// EPI 1: f16 TRANSPOSED store [n][M] + bias[m] (conv2 -> h2t, pre-BN)
// EPI 2: fused stats atomics + maxpool + bias[m] (conv3)
template<int EPI>
__global__ void gemm_f16(const _Float16* __restrict__ A16, const _Float16* __restrict__ B16,
                         const float* __restrict__ bias,
                         void* __restrict__ Cout,
                         int M, int N, int K,
                         long strideB, long strideC,
                         float* __restrict__ sumBuf, float* __restrict__ sumsqBuf,
                         float* __restrict__ pooledBuf) {
    int wave = threadIdx.x >> 5;
    int lane = threadIdx.x & 31;
    int half = lane >> 4;
    int l16  = lane & 15;
    int mtile = blockIdx.y * 8 + wave;
    if (mtile * 16 >= M) return;
    int m0 = mtile * 16;
    int nb = blockIdx.x * 64;
    int bz = blockIdx.z;
    const _Float16* Bb = B16 + (long)bz * strideB;
    const _Float16* ap = A16 + (long)(m0 + l16) * K + 8 * half;

    v8f acc[4] = {{}, {}, {}, {}};
    for (int k0 = 0; k0 < K; k0 += 32) {
        v16h af;
        {
            v8h lo = *(const v8h*)(ap + k0);
            v8h hi = *(const v8h*)(ap + k0 + 16);
#pragma unroll
            for (int i = 0; i < 8; ++i) { af[i] = lo[i]; af[8 + i] = hi[i]; }
        }
#pragma unroll
        for (int q = 0; q < 4; ++q) {
            const _Float16* bp = Bb + (long)(nb + q * 16 + l16) * K + k0 + 8 * half;
            v8h blo = *(const v8h*)bp;
            v8h bhi = *(const v8h*)(bp + 16);
            v16h bf;
#pragma unroll
            for (int i = 0; i < 8; ++i) { bf[i] = blo[i]; bf[8 + i] = bhi[i]; }
            acc[q] = __builtin_amdgcn_wmma_f32_16x16x32_f16(
                false, af, false, bf, (short)0, acc[q], false, false);
        }
    }

    if (EPI == 0) {              // f32 C[M,N], bias per n
        float* C = (float*)Cout;
#pragma unroll
        for (int q = 0; q < 4; ++q) {
            int n = nb + q * 16 + l16;
#pragma unroll
            for (int r = 0; r < 8; ++r) {
                int row = m0 + r + 8 * half;
                C[(long)row * N + n] = acc[q][r] + bias[n];
            }
        }
    } else if (EPI == 1) {       // f16 transposed [n][M], bias per m (pre-BN)
        _Float16* C = (_Float16*)Cout + (long)bz * strideC;
#pragma unroll
        for (int q = 0; q < 4; ++q) {
            int n = nb + q * 16 + l16;
            v8h outv;
#pragma unroll
            for (int r = 0; r < 8; ++r)
                outv[r] = (_Float16)(acc[q][r] + bias[m0 + r + 8 * half]);
            *(v8h*)((_Float16*)C + (long)n * M + m0 + 8 * half) = outv;
        }
    } else {                     // conv3: stats atomics + maxpool, bias per m
#pragma unroll
        for (int q = 0; q < 4; ++q) {
            int cg = (nb + q * 16) >> 5;   // 16-wide subtile inside one 32-wide c group
#pragma unroll
            for (int r = 0; r < 8; ++r) {
                int row = m0 + r + 8 * half;
                float v = acc[q][r] + bias[row];
                float s = v, ss = v * v, mx = v;
#pragma unroll
                for (int d = 1; d < 16; d <<= 1) {
                    s  += __shfl_xor(s, d);
                    ss += __shfl_xor(ss, d);
                    mx  = fmaxf(mx, __shfl_xor(mx, d));
                }
                if (l16 == 0) {
                    atomicAdd(&sumBuf[row], s);
                    atomicAdd(&sumsqBuf[row], ss);
                    atomicMaxF(&pooledBuf[((long)bz * 256 + row) * NCENT + cg], mx);
                }
            }
        }
    }
}

// ---------------- finalize bn3 ----------------------------------------------
__global__ void finalize_bn3(const float* __restrict__ sums,
                             float* __restrict__ mean, float* __restrict__ inv) {
    int ch = blockIdx.x * 256 + threadIdx.x;
    if (ch < 256) {
        float cnt = 131072.f;   // B * C * K
        float m = sums[ch] / cnt;
        float v = sums[256 + ch] / cnt - m * m;
        mean[ch] = m;
        inv[ch] = rsqrtf(fmaxf(v, 0.f) + BNEPS);
    }
}

// ---------------- BN3 + relu on pooled maxes -> f16 FC input ----------------
__global__ void pool_bn_kernel(const float* __restrict__ pooled,
                               const float* __restrict__ m, const float* __restrict__ inv,
                               const float* __restrict__ g3, const float* __restrict__ b3,
                               _Float16* __restrict__ fcin16) {
    int i = blockIdx.x * 256 + threadIdx.x;   // B*8192
    int o = (i & 8191) >> 5;
    float v = pooled[i];
    fcin16[i] = (_Float16)fmaxf(0.f, (v - m[o]) * inv[o] * g3[o] + b3[o]);
}

// ---------------- tiny tail: fc3 / vae / decoder, one block per batch -------
__global__ void tail_kernel(const float* __restrict__ fc2out,
                            const float* __restrict__ bn5m, const float* __restrict__ bn5i,
                            const float* __restrict__ g5, const float* __restrict__ b5,
                            const float* __restrict__ fc3w, const float* __restrict__ fc3b,
                            const float* __restrict__ fce1w, const float* __restrict__ fce1b,
                            const float* __restrict__ fce2w, const float* __restrict__ fce2b,
                            const float* __restrict__ eps,
                            const float* __restrict__ dfc1w, const float* __restrict__ dfc1b,
                            const float* __restrict__ dfc2w, const float* __restrict__ dfc2b,
                            const float* __restrict__ dfc3w, const float* __restrict__ dfc3b,
                            float* __restrict__ out) {
    int b = blockIdx.x, t = threadIdx.x;   // 256 threads
    __shared__ float h[256], feat[8], z[8], d1[32], d2[128];
    float v = fc2out[b * 256 + t];
    h[t] = fmaxf(0.f, (v - bn5m[t]) * bn5i[t] * g5[t] + b5[t]);
    __syncthreads();
    if (t < 8) {
        float a = fc3b[t];
        for (int k = 0; k < 256; ++k) a += fc3w[t * 256 + k] * h[k];
        feat[t] = fmaxf(a, 0.f);
    }
    __syncthreads();
    if (t < 8) {
        float mu = fce1b[t], lv = fce2b[t];
        for (int k = 0; k < 8; ++k) {
            mu += fce1w[t * 8 + k] * feat[k];
            lv += fce2w[t * 8 + k] * feat[k];
        }
        z[t] = eps[b * 8 + t] * expf(0.5f * lv) + mu;
        out[65536 + b * 8 + t] = mu;
        out[66560 + b * 8 + t] = lv;
    }
    __syncthreads();
    if (t < 32) {
        float a = dfc1b[t];
        for (int k = 0; k < 8; ++k) a += dfc1w[t * 8 + k] * z[k];
        d1[t] = fmaxf(a, 0.f);
    }
    __syncthreads();
    if (t < 128) {
        float a = dfc2b[t];
        for (int k = 0; k < 32; ++k) a += dfc2w[t * 32 + k] * d1[k];
        d2[t] = fmaxf(a, 0.f);
    }
    __syncthreads();
    for (int o = t; o < 512; o += 256) {
        float a = dfc3b[o];
        for (int k = 0; k < 128; ++k) a += dfc3w[o * 128 + k] * d2[k];
        out[b * 512 + o] = a;
    }
}

// ===========================================================================
extern "C" void kernel_launch(void* const* d_in, const int* in_sizes, int n_in,
                              void* d_out, int out_size, void* d_ws, size_t ws_size,
                              hipStream_t stream) {
    const float* x      = (const float*)d_in[0];
    const float* eps    = (const float*)d_in[1];
    const float* c1w    = (const float*)d_in[2];
    const float* c1b    = (const float*)d_in[3];
    const float* bn1g   = (const float*)d_in[4];
    const float* bn1b   = (const float*)d_in[5];
    const float* c2w    = (const float*)d_in[6];
    const float* c2b    = (const float*)d_in[7];
    const float* bn2g   = (const float*)d_in[8];
    const float* bn2b   = (const float*)d_in[9];
    const float* c3w    = (const float*)d_in[10];
    const float* c3b    = (const float*)d_in[11];
    const float* bn3g   = (const float*)d_in[12];
    const float* bn3b   = (const float*)d_in[13];
    const float* fc1w   = (const float*)d_in[14];
    const float* fc1b   = (const float*)d_in[15];
    const float* bn4g   = (const float*)d_in[16];
    const float* bn4b   = (const float*)d_in[17];
    const float* fc2w   = (const float*)d_in[18];
    const float* fc2b   = (const float*)d_in[19];
    const float* bn5g   = (const float*)d_in[20];
    const float* bn5b   = (const float*)d_in[21];
    const float* fc3w   = (const float*)d_in[22];
    const float* fc3b   = (const float*)d_in[23];
    const float* fce1w  = (const float*)d_in[24];
    const float* fce1b  = (const float*)d_in[25];
    const float* fce2w  = (const float*)d_in[26];
    const float* fce2b  = (const float*)d_in[27];
    const float* dfc1w  = (const float*)d_in[28];
    const float* dfc1b  = (const float*)d_in[29];
    const float* dfc2w  = (const float*)d_in[30];
    const float* dfc2b  = (const float*)d_in[31];
    const float* dfc3w  = (const float*)d_in[32];
    const float* dfc3b  = (const float*)d_in[33];

    float* ws = (float*)d_ws;
    float*     cent   = ws + OFF_CENT;
    float*     g      = ws + OFF_G;
    _Float16*  h1t    = (_Float16*)(ws + OFF_H1T);
    _Float16*  h2t    = (_Float16*)(ws + OFF_H2T);
    float*     pool   = ws + OFF_POOL;
    _Float16*  fcin16 = (_Float16*)(ws + OFF_FCIN);
    float*     fc1o   = ws + OFF_FC1;
    _Float16*  fc1p16 = (_Float16*)(ws + OFF_FC1P);
    float*     fc2o   = ws + OFF_FC2;
    _Float16*  c2w16  = (_Float16*)(ws + OFF_W2);
    _Float16*  c3w16  = (_Float16*)(ws + OFF_W3);
    _Float16*  fc2w16 = (_Float16*)(ws + OFF_WF2);
    _Float16*  fc1w16 = (_Float16*)(ws + OFF_WF1);
    float* bn1m = ws + OFF_BN1M; float* bn1i = ws + OFF_BN1I;
    float* bn2m = ws + OFF_BN2M; float* bn2i = ws + OFF_BN2I;
    float* bn3s = ws + OFF_BN3S;
    float* bn3m = ws + OFF_BN3M; float* bn3i = ws + OFF_BN3I;
    float* bn4m = ws + OFF_BN4M; float* bn4i = ws + OFF_BN4I;
    float* bn5m = ws + OFF_BN5M; float* bn5i = ws + OFF_BN5I;
    float* outp = (float*)d_out;

    // 0) init pool(-inf) + bn3 accumulators
    init_kernel<<<(BATCH * 256 * NCENT + 255) / 256, 256, 0, stream>>>(pool, bn3s);
    // 1) FPS
    fps_kernel<<<BATCH, 256, 0, stream>>>(x, cent);
    // 2) grouping
    group_kernel<<<dim3(NCENT, BATCH), 32, 0, stream>>>(x, cent, g);
    // 3) conv1 -> pre-BN f16 transposed h1t [b][ck][64]
    conv1t_kernel<<<dim3(4, BATCH), 256, 0, stream>>>(g, c1w, c1b, h1t);
    // 4) bn1 stats (f16 buffer, stride 64, 131072 elems/channel)
    stats16_kernel<<<64, 256, 0, stream>>>(h1t, bn1m, bn1i, BATCH * 1024, 64);
    // 5) in-place BN1+ReLU on h1t
    bnrelu16_kernel<<<32768, 256, 0, stream>>>(h1t, bn1m, bn1i, bn1g, bn1b,
                                               63, 8388608L);
    // 6) convert weights to f16
    cvt_kernel<<<32, 256, 0, stream>>>(c2w, c2w16, 8192L);
    cvt_kernel<<<128, 256, 0, stream>>>(c3w, c3w16, 32768L);
    cvt_kernel<<<1024, 256, 0, stream>>>(fc2w, fc2w16, 262144L);
    cvt_kernel<<<32768, 256, 0, stream>>>(fc1w, fc1w16, 8388608L);
    // 7) conv2 WMMA: h2t(pre-BN f16, transposed) = c2w16 x h1t^T, per batch
    gemm_f16<1><<<dim3(16, 1, BATCH), 256, 0, stream>>>(
        c2w16, h1t, c2b, h2t, 128, 1024, 64,
        65536L /*1024*64 halves*/, 131072L /*1024*128 halves*/,
        nullptr, nullptr, nullptr);
    // 8) bn2 stats
    stats16_kernel<<<128, 256, 0, stream>>>(h2t, bn2m, bn2i, BATCH * 1024, 128);
    // 9) in-place BN2+ReLU on h2t
    bnrelu16_kernel<<<65536, 256, 0, stream>>>(h2t, bn2m, bn2i, bn2g, bn2b,
                                               127, 16777216L);
    // 10) conv3 WMMA fused (stats atomics + maxpool), M=256 K=128 N=1024
    gemm_f16<2><<<dim3(16, 2, BATCH), 256, 0, stream>>>(
        c3w16, h2t, c3b, nullptr, 256, 1024, 128,
        131072L, 0, bn3s, bn3s + 256, pool);
    // 11) finalize bn3
    finalize_bn3<<<1, 256, 0, stream>>>(bn3s, bn3m, bn3i);
    // 12) bn3+relu on pooled -> fcin16 (B x 8192, f16)
    pool_bn_kernel<<<(BATCH * 8192) / 256, 256, 0, stream>>>(pool, bn3m, bn3i, bn3g, bn3b, fcin16);
    // 13) fc1 WMMA: M=128 K=8192 N=1024
    gemm_f16<0><<<dim3(16, 1, 1), 256, 0, stream>>>(
        fcin16, fc1w16, fc1b, fc1o, 128, 1024, 8192,
        0, 0, nullptr, nullptr, nullptr);
    // 14) bn4 stats (f32, per column of (128,1024))
    stats32_kernel<<<1024, 256, 0, stream>>>(fc1o, bn4m, bn4i, BATCH, 1024);
    // 15) fc1o -> f16 with BN4+ReLU
    cvtbn_kernel<<<512, 256, 0, stream>>>(fc1o, fc1p16, bn4m, bn4i, bn4g, bn4b,
                                          1023, 131072L);
    // 16) fc2 WMMA: M=128 K=1024 N=256
    gemm_f16<0><<<dim3(4, 1, 1), 256, 0, stream>>>(
        fc1p16, fc2w16, fc2b, fc2o, 128, 256, 1024,
        0, 0, nullptr, nullptr, nullptr);
    // 17) bn5 stats
    stats32_kernel<<<256, 256, 0, stream>>>(fc2o, bn5m, bn5i, BATCH, 256);
    // 18) tail: fc3 + VAE + decoder -> d_out
    tail_kernel<<<BATCH, 256, 0, stream>>>(
        fc2o, bn5m, bn5i, bn5g, bn5b, fc3w, fc3b,
        fce1w, fce1b, fce2w, fce2b, eps,
        dfc1w, dfc1b, dfc2w, dfc2b, dfc3w, dfc3b, outp);
}
